// RealImagSelfAttention_17978733101521
// MI455X (gfx1250) — matrix operand order repaired and checked by reading, not compiled, not run on hardware
//
#include <hip/hip_runtime.h>

// ---------------------------------------------------------------------------
// Types for CDNA5 WMMA (wave32, v_wmma_f32_16x16x32_bf16)
// ---------------------------------------------------------------------------
typedef __attribute__((ext_vector_type(16))) __bf16 v16bf;
typedef __attribute__((ext_vector_type(8)))  __bf16 v8bf;
typedef __attribute__((ext_vector_type(8)))  float  v8f;

#define B_SZ 2
#define S_SZ 2048
#define D_SZ 2048
#define H_N  16
#define HD_N 128
#define M_TOT 4096          // B*S
#define K2    4096          // 2*D (stacked real|imag)
#define N_QKV 12288         // 6*D

// ---- optional CDNA5 async global->LDS path (ASYNCcnt-tracked) -------------
#if defined(__HIP_DEVICE_COMPILE__) && defined(__has_builtin)
#if __has_builtin(__builtin_amdgcn_global_load_async_to_lds_b128) && \
    __has_builtin(__builtin_amdgcn_s_wait_asynccnt)
#define USE_ASYNC_LDS 1
#endif
#endif
#ifndef USE_ASYNC_LDS
#define USE_ASYNC_LDS 0
#endif

#if USE_ASYNC_LDS
typedef int v4i_vec __attribute__((vector_size(16)));
typedef __attribute__((address_space(1))) v4i_vec* gptr_v4i;
typedef __attribute__((address_space(3))) v4i_vec* lptr_v4i;
__device__ __forceinline__ void async_copy16(const void* g, void* l) {
    __builtin_amdgcn_global_load_async_to_lds_b128(
        (gptr_v4i)g, (lptr_v4i)l, 0, 0);
}
__device__ __forceinline__ void async_wait() {
    __builtin_amdgcn_s_wait_asynccnt(0);
}
#else
__device__ __forceinline__ void async_wait() {}
#endif

// Build a v16bf fragment from two 16-byte-aligned groups of 8 bf16.
__device__ __forceinline__ v16bf make16(const __bf16* p0, const __bf16* p1) {
    v8bf a = *(const v8bf*)p0;
    v8bf b = *(const v8bf*)p1;
    v16bf r;
#pragma unroll
    for (int i = 0; i < 8; ++i) { r[i] = a[i]; r[i + 8] = b[i]; }
    return r;
}

// ---------------------------------------------------------------------------
// Pack kernels (fp32 -> bf16, building K-stacked operands)
// ---------------------------------------------------------------------------
__global__ void pack_x_kernel(const float* __restrict__ xr,
                              const float* __restrict__ xi,
                              __bf16* __restrict__ dst) {
    size_t i = (size_t)blockIdx.x * 256 + threadIdx.x;   // over 4096*4096
    int m = (int)(i >> 12);
    int k = (int)(i & 4095);
    float v = (k < D_SZ) ? xr[(size_t)m * D_SZ + k]
                         : xi[(size_t)m * D_SZ + (k - D_SZ)];
    dst[i] = (__bf16)v;
}

// dst[r][colOff + c] = r < 2048 ? top[r][c] : sgnBot * bot[r-2048][c]
__global__ void pack_w_kernel(const float* __restrict__ top,
                              const float* __restrict__ bot, float sgnBot,
                              __bf16* __restrict__ dst, int ldDst, int colOff) {
    size_t i = (size_t)blockIdx.x * 256 + threadIdx.x;   // over 4096*2048
    int r = (int)(i >> 11);
    int c = (int)(i & 2047);
    float v = (r < D_SZ) ? top[(size_t)r * D_SZ + c]
                         : sgnBot * bot[(size_t)(r - D_SZ) * D_SZ + c];
    dst[(size_t)r * ldDst + colOff + c] = (__bf16)v;
}

// ---------------------------------------------------------------------------
// Tiled bf16 WMMA GEMM: C[M,N] = A[M,K] * W[K,N].
// 128x128 block, 8 waves (4x2), 32x64 per wave (2x4 fragments).
// MODE 0: bf16 output, row-major ld=N.  MODE 1: f32, split halves into d_out.
// ---------------------------------------------------------------------------
template <int MODE>
__global__ void __launch_bounds__(256)
gemm_bf16(const __bf16* __restrict__ A, const __bf16* __restrict__ W,
          void* __restrict__ Cout, int N, int K, size_t halfStride) {
    __shared__ __align__(16) __bf16 As[128][72];   // K-major, +8 pad (144B row)
    __shared__ __align__(16) __bf16 Bt[128][72];   // N-major (transposed), +8 pad

    const int tid  = threadIdx.x;
    const int lane = tid & 31;
    const int wave = tid >> 5;
    const int wm   = wave >> 1;      // 0..3
    const int wn   = wave & 1;       // 0..1
    const int lh   = lane >> 4;      // 0/1 (K-split half per ISA layout)
    const int ll   = lane & 15;
    const int m0   = blockIdx.y * 128;
    const int n0   = blockIdx.x * 128;

    v8f acc[2][4] = {};

    for (int kt = 0; kt < K; kt += 64) {
        // ---- A tile: 128x64 bf16 = 1024 x 16B chunks, 4 per thread ----
#if USE_ASYNC_LDS
#pragma unroll
        for (int q = 0; q < 4; ++q) {
            int lin = tid + q * 256;
            int r = lin >> 3;
            int c = (lin & 7) << 3;
            async_copy16(A + (size_t)(m0 + r) * K + kt + c, &As[r][c]);
        }
#else
        {
            v8bf ta[4];
#pragma unroll
            for (int q = 0; q < 4; ++q) {
                int lin = tid + q * 256;
                int r = lin >> 3;
                int c = (lin & 7) << 3;
                ta[q] = *(const v8bf*)(A + (size_t)(m0 + r) * K + kt + c);
            }
#pragma unroll
            for (int q = 0; q < 4; ++q) {
                int lin = tid + q * 256;
                int r = lin >> 3;
                int c = (lin & 7) << 3;
                *(v8bf*)(&As[r][c]) = ta[q];
            }
        }
#endif
        // ---- B tile: 64x128 loaded row-major, stored transposed ----
        {
            v8bf tb[4];
#pragma unroll
            for (int q = 0; q < 4; ++q) {
                int lin = tid + q * 256;
                int kk = lin >> 4;
                int c  = (lin & 15) << 3;
                tb[q] = *(const v8bf*)(W + (size_t)(kt + kk) * N + n0 + c);
            }
#pragma unroll
            for (int q = 0; q < 4; ++q) {
                int lin = tid + q * 256;
                int kk = lin >> 4;
                int c  = (lin & 15) << 3;
#pragma unroll
                for (int j = 0; j < 8; ++j) Bt[c + j][kk] = tb[q][j];
            }
        }
        // prefetch next k-tile into cache while this one is consumed
        if (kt + 64 < K) {
            __builtin_prefetch(A + (size_t)(m0 + (tid >> 1)) * K + kt + 64 +
                                   (tid & 1) * 32, 0, 1);
            __builtin_prefetch(W + (size_t)(kt + 64 + (tid >> 2)) * N + n0 +
                                   (tid & 3) * 32, 0, 1);
        }
        async_wait();
        __syncthreads();

#pragma unroll
        for (int ks = 0; ks < 2; ++ks) {
            const int kb = ks * 32;
            v16bf af[2], bfr[4];
#pragma unroll
            for (int mf = 0; mf < 2; ++mf) {
                int r = wm * 32 + mf * 16 + ll;
                const __bf16* p0 = &As[r][kb + lh * 8];
                af[mf] = make16(p0, p0 + 16);
            }
#pragma unroll
            for (int nf = 0; nf < 4; ++nf) {
                int cN = wn * 64 + nf * 16 + ll;
                const __bf16* p0 = &Bt[cN][kb + lh * 8];
                bfr[nf] = make16(p0, p0 + 16);
            }
#pragma unroll
            for (int mf = 0; mf < 2; ++mf)
#pragma unroll
                for (int nf = 0; nf < 4; ++nf)
                    acc[mf][nf] = __builtin_amdgcn_wmma_f32_16x16x32_bf16(
                        false, af[mf], false, bfr[nf], (short)0, acc[mf][nf],
                        false, false);
        }
        __syncthreads();
    }

#pragma unroll
    for (int mf = 0; mf < 2; ++mf)
#pragma unroll
        for (int nf = 0; nf < 4; ++nf)
#pragma unroll
            for (int v = 0; v < 8; ++v) {
                int r = m0 + wm * 32 + mf * 16 + v + lh * 8;
                int c = n0 + wn * 64 + nf * 16 + ll;
                float val = acc[mf][nf][v];
                if (MODE == 0) {
                    ((__bf16*)Cout)[(size_t)r * N + c] = (__bf16)val;
                } else {
                    float* o = (float*)Cout;
                    if (c < D_SZ) o[(size_t)r * D_SZ + c] = val;
                    else o[halfStride + (size_t)r * D_SZ + (c - D_SZ)] = val;
                }
            }
}

// ---------------------------------------------------------------------------
// RoPE + rearrange to head layout [B,H,S,256] (r|i stacked), Q pre-scaled.
// ---------------------------------------------------------------------------
__global__ void rope_kernel(const __bf16* __restrict__ qkv,
                            __bf16* __restrict__ Qh, __bf16* __restrict__ Kh,
                            __bf16* __restrict__ Vh) {
    size_t i = (size_t)blockIdx.x * 256 + threadIdx.x;  // B*S*H*64
    int j = (int)(i & 63);
    int h = (int)((i >> 6) & 15);
    int s = (int)((i >> 10) & 2047);
    int b = (int)(i >> 21);

    const __bf16* row = qkv + (size_t)(b * S_SZ + s) * N_QKV;
    size_t hb = (((size_t)(b * H_N + h)) * S_SZ + s) * 256;

    float inv = __powf(10000.f, -(float)j * (1.f / 64.f));
    float ang = (float)s * inv;
    float c, sn;
    __sincosf(ang, &c, &sn);
    const float qs = 0.08838834764831845f;  // 1/sqrt(HD)

#pragma unroll
    for (int g = 0; g < 4; ++g) {  // 0:q_r 1:q_i 2:k_r 3:k_i
        int base = g * D_SZ + h * HD_N;
        float x1 = (float)row[base + j];
        float x2 = (float)row[base + 64 + j];
        float o1 = x1 * c - x2 * sn;
        float o2 = x2 * c + x1 * sn;
        int doff = (g & 1) * HD_N;
        if (g < 2) {
            Qh[hb + doff + j]      = (__bf16)(o1 * qs);
            Qh[hb + doff + 64 + j] = (__bf16)(o2 * qs);
        } else {
            Kh[hb + doff + j]      = (__bf16)o1;
            Kh[hb + doff + 64 + j] = (__bf16)o2;
        }
    }
    // V: plain rearrange
    int vr = 4 * D_SZ + h * HD_N;
    int vi = 5 * D_SZ + h * HD_N;
    Vh[hb + j]        = row[vr + j];
    Vh[hb + 64 + j]   = row[vr + 64 + j];
    Vh[hb + 128 + j]  = row[vi + j];
    Vh[hb + 192 + j]  = row[vi + 64 + j];
}

// ---------------------------------------------------------------------------
// Causal flash attention, d_qk = d_v = 256 (stacked r|i), WMMA everywhere.
// 4 waves / WG; 64-query tile (16 rows per wave); 64-key tiles.
// ---------------------------------------------------------------------------
__global__ void __launch_bounds__(128)
attn_kernel(const __bf16* __restrict__ Qh, const __bf16* __restrict__ Kh,
            const __bf16* __restrict__ Vh, __bf16* __restrict__ attnA) {
    // K tile (64x264) and V^T tile (256x72) share one buffer (phased use)
    __shared__ __align__(16) __bf16 KV[256 * 72];        // 36,864 B
    __shared__ __align__(16) __bf16 Pt[4][16][72];       //  9,216 B
    __bf16 (*Kt)[264] = (__bf16(*)[264])KV;
    __bf16 (*Vt)[72]  = (__bf16(*)[72])KV;

    const int tid  = threadIdx.x;
    const int lane = tid & 31;
    const int w    = tid >> 5;
    const int lh   = lane >> 4;
    const int ll   = lane & 15;

    const int blk = blockIdx.x;           // b*512 + h*32 + qt
    const int qt  = blk & 31;
    const int h   = (blk >> 5) & 15;
    const int b   = blk >> 9;
    const int Q0  = qt * 64;
    const size_t headBase = ((size_t)(b * H_N + h)) * S_SZ * 256;

    // Q fragments resident in registers (16 rows x 256)
    v16bf qf[8];
    {
        const __bf16* qp = Qh + headBase + (size_t)(Q0 + w * 16 + ll) * 256;
#pragma unroll
        for (int kk = 0; kk < 8; ++kk) {
            const __bf16* p0 = qp + kk * 32 + lh * 8;
            qf[kk] = make16(p0, p0 + 16);
        }
    }

    v8f O[16] = {};
    float mrow[8], lrow[8];
#pragma unroll
    for (int v = 0; v < 8; ++v) { mrow[v] = -3.0e38f; lrow[v] = 0.f; }

    for (int kv = 0; kv <= Q0; kv += 64) {
        const __bf16* kg = Kh + headBase + (size_t)kv * 256;
        const __bf16* vg = Vh + headBase + (size_t)kv * 256;

        // ---- load K tile row-major [64][256] (+pad) ----
#if USE_ASYNC_LDS
#pragma unroll
        for (int q = 0; q < 16; ++q) {
            int lin = tid + q * 128;          // 0..2047
            int r = lin >> 5;
            int c = (lin & 31) << 3;
            async_copy16(kg + (size_t)r * 256 + c, &Kt[r][c]);
        }
        async_wait();
#else
        {
            v8bf tk[16];
#pragma unroll
            for (int q = 0; q < 16; ++q) {
                int lin = tid + q * 128;
                int r = lin >> 5;
                int c = (lin & 31) << 3;
                tk[q] = *(const v8bf*)(kg + (size_t)r * 256 + c);
            }
#pragma unroll
            for (int q = 0; q < 16; ++q) {
                int lin = tid + q * 128;
                int r = lin >> 5;
                int c = (lin & 31) << 3;
                *(v8bf*)(&Kt[r][c]) = tk[q];
            }
        }
#endif
        __syncthreads();

        // ---- scores S = Q * K^T (16x64 per wave) ----
        v8f S[4] = {};
#pragma unroll
        for (int nf = 0; nf < 4; ++nf) {
            int cN = nf * 16 + ll;
#pragma unroll
            for (int kk = 0; kk < 8; ++kk) {
                const __bf16* p0 = &Kt[cN][kk * 32 + lh * 8];
                v16bf bfr = make16(p0, p0 + 16);
                S[nf] = __builtin_amdgcn_wmma_f32_16x16x32_bf16(
                    false, qf[kk], false, bfr, (short)0, S[nf], false, false);
            }
        }

        // ---- online softmax ----
#pragma unroll
        for (int v = 0; v < 8; ++v) {
            int qg = Q0 + w * 16 + v + lh * 8;
            float mx = -3.0e38f;
#pragma unroll
            for (int nf = 0; nf < 4; ++nf) {
                int kgi = kv + nf * 16 + ll;
                float s = S[nf][v];
                if (kgi > qg) s = -3.0e38f;
                S[nf][v] = s;
                mx = fmaxf(mx, s);
            }
#pragma unroll
            for (int off = 8; off >= 1; off >>= 1)
                mx = fmaxf(mx, __shfl_xor(mx, off, 32));
            float mn = fmaxf(mrow[v], mx);
            float corr = __expf(mrow[v] - mn);
            mrow[v] = mn;
            float rs = 0.f;
#pragma unroll
            for (int nf = 0; nf < 4; ++nf) {
                float p = __expf(S[nf][v] - mn);
                S[nf][v] = p;
                rs += p;
            }
#pragma unroll
            for (int off = 8; off >= 1; off >>= 1)
                rs += __shfl_xor(rs, off, 32);
            lrow[v] = lrow[v] * corr + rs;
#pragma unroll
            for (int nf2 = 0; nf2 < 16; ++nf2) O[nf2][v] *= corr;
#pragma unroll
            for (int nf = 0; nf < 4; ++nf)
                Pt[w][v + lh * 8][nf * 16 + ll] = (__bf16)S[nf][v];
        }
        __syncthreads();   // all waves done with Kt; Pt visible

        // ---- load V tile transposed [256][64] (+pad) over same buffer ----
        {
            v8bf tv[16];
#pragma unroll
            for (int q = 0; q < 16; ++q) {
                int lin = tid + q * 128;
                int r = lin >> 5;
                int c = (lin & 31) << 3;
                tv[q] = *(const v8bf*)(vg + (size_t)r * 256 + c);
            }
#pragma unroll
            for (int q = 0; q < 16; ++q) {
                int lin = tid + q * 128;
                int r = lin >> 5;
                int c = (lin & 31) << 3;
#pragma unroll
                for (int j = 0; j < 8; ++j) Vt[c + j][r] = tv[q][j];
            }
        }
        __syncthreads();

        // ---- O += P * V (16x256 per wave) ----
        v16bf pf[2];
#pragma unroll
        for (int kk = 0; kk < 2; ++kk) {
            const __bf16* p0 = &Pt[w][ll][kk * 32 + lh * 8];
            pf[kk] = make16(p0, p0 + 16);
        }
#pragma unroll
        for (int nf2 = 0; nf2 < 16; ++nf2) {
            int cN = nf2 * 16 + ll;
#pragma unroll
            for (int kk = 0; kk < 2; ++kk) {
                const __bf16* p0 = &Vt[cN][kk * 32 + lh * 8];
                v16bf bfr = make16(p0, p0 + 16);
                O[nf2] = __builtin_amdgcn_wmma_f32_16x16x32_bf16(
                    false, pf[kk], false, bfr, (short)0, O[nf2], false, false);
            }
        }
        __syncthreads();   // before next tile overwrites KV buffer
    }

    // ---- normalize & write to K-stacked attention output [4096][4096] ----
#pragma unroll
    for (int nf2 = 0; nf2 < 16; ++nf2) {
        int c = nf2 * 16 + ll;
        int gcol = (c < HD_N) ? (h * HD_N + c) : (D_SZ + h * HD_N + (c - HD_N));
#pragma unroll
        for (int v = 0; v < 8; ++v) {
            int qg = Q0 + w * 16 + v + lh * 8;
            float val = O[nf2][v] / lrow[v];
            attnA[((size_t)(b * S_SZ + qg)) * K2 + gcol] = (__bf16)val;
        }
    }
}

// ---------------------------------------------------------------------------
// Host launcher
// ---------------------------------------------------------------------------
extern "C" void kernel_launch(void* const* d_in, const int* in_sizes, int n_in,
                              void* d_out, int out_size, void* d_ws,
                              size_t ws_size, hipStream_t stream) {
    (void)in_sizes; (void)n_in; (void)out_size; (void)ws_size;
    const float* xr   = (const float*)d_in[0];
    const float* xi   = (const float*)d_in[1];
    const float* Wq_r = (const float*)d_in[2];
    const float* Wq_i = (const float*)d_in[3];
    const float* Wk_r = (const float*)d_in[4];
    const float* Wk_i = (const float*)d_in[5];
    const float* Wv_r = (const float*)d_in[6];
    const float* Wv_i = (const float*)d_in[7];
    const float* Wo_r = (const float*)d_in[8];
    const float* Wo_i = (const float*)d_in[9];

    char* ws = (char*)d_ws;
    __bf16* Ast   = (__bf16*)(ws + 0);                 //  33.5 MB [4096,4096]
    __bf16* Wqkv  = (__bf16*)(ws + 33554432ull);       // 100.7 MB [4096,12288]
    __bf16* QKV   = (__bf16*)(ws + 134217728ull);      // 100.7 MB [4096,12288]
    __bf16* Qh    = (__bf16*)(ws + 234881024ull);      //  33.5 MB [B,H,S,256]
    __bf16* Kh    = (__bf16*)(ws + 268435456ull);
    __bf16* Vh    = (__bf16*)(ws + 301989888ull);
    __bf16* AttnA = QKV;   // reuse after rope consumed QKV
    __bf16* Wo    = Ast;   // reuse after QKV GEMM consumed Ast

    // 1) pack inputs + QKV stacked weights
    pack_x_kernel<<<65536, 256, 0, stream>>>(xr, xi, Ast);
    pack_w_kernel<<<32768, 256, 0, stream>>>(Wq_r, Wq_i, -1.f, Wqkv, N_QKV, 0);
    pack_w_kernel<<<32768, 256, 0, stream>>>(Wq_i, Wq_r,  1.f, Wqkv, N_QKV, 2048);
    pack_w_kernel<<<32768, 256, 0, stream>>>(Wk_r, Wk_i, -1.f, Wqkv, N_QKV, 4096);
    pack_w_kernel<<<32768, 256, 0, stream>>>(Wk_i, Wk_r,  1.f, Wqkv, N_QKV, 6144);
    pack_w_kernel<<<32768, 256, 0, stream>>>(Wv_r, Wv_i, -1.f, Wqkv, N_QKV, 8192);
    pack_w_kernel<<<32768, 256, 0, stream>>>(Wv_i, Wv_r,  1.f, Wqkv, N_QKV, 10240);

    // 2) QKV projection: [4096,4096] x [4096,12288]
    gemm_bf16<0><<<dim3(N_QKV / 128, M_TOT / 128), 256, 0, stream>>>(
        Ast, Wqkv, (void*)QKV, N_QKV, K2, 0);

    // 3) RoPE + head rearrange
    rope_kernel<<<16384, 256, 0, stream>>>(QKV, Qh, Kh, Vh);

    // 4) pack output-projection stacked weights (reuses Ast region)
    pack_w_kernel<<<32768, 256, 0, stream>>>(Wo_r, Wo_i, -1.f, Wo, K2, 0);
    pack_w_kernel<<<32768, 256, 0, stream>>>(Wo_i, Wo_r,  1.f, Wo, K2, 2048);

    // 5) causal flash attention (WMMA), writes K-stacked [4096,4096] bf16
    attn_kernel<<<B_SZ * H_N * (S_SZ / 64), 128, 0, stream>>>(Qh, Kh, Vh, AttnA);

    // 6) output projection -> f32 d_out (out_r then out_i)
    gemm_bf16<1><<<dim3(K2 / 128, M_TOT / 128), 256, 0, stream>>>(
        AttnA, Wo, d_out, K2, K2, (size_t)M_TOT * D_SZ);
}
